// MultiHeadSelfAttention_37211596652859
// MI455X (gfx1250) — matrix operand order repaired and checked by reading, not compiled, not run on hardware
//
#include <hip/hip_runtime.h>
#include <hip/hip_bf16.h>

#define USE_TDM 1

typedef __bf16 bf16_t;
typedef __attribute__((ext_vector_type(16))) __bf16 v16bf;
typedef __attribute__((ext_vector_type(8)))  __bf16 v8bf;
typedef __attribute__((ext_vector_type(8)))  float  v8f;
typedef __attribute__((ext_vector_type(4)))  unsigned u32x4;
typedef __attribute__((ext_vector_type(8)))  unsigned u32x8;

#define DMODEL 1024
#define NHEADS 16
#define DK     64
#define BATCH  4
#define SEQ    2048
#define MROWS  (BATCH * SEQ)   // 8192

static __device__ __forceinline__ unsigned short f2bf_bits(float f) {
    union { float f; unsigned u; } x; x.f = f;
    unsigned r = x.u + 0x7FFFu + ((x.u >> 16) & 1u);   // RNE
    return (unsigned short)(r >> 16);
}
static __device__ __forceinline__ bf16_t f2bf(float f) {
    union { unsigned short s; bf16_t b; } y; y.s = f2bf_bits(f); return y.b;
}
static __device__ __forceinline__ v16bf load_frag(const bf16_t* p0, const bf16_t* p1) {
    v8bf lo = *reinterpret_cast<const v8bf*>(p0);
    v8bf hi = *reinterpret_cast<const v8bf*>(p1);
    v16bf r;
#pragma unroll
    for (int i = 0; i < 8; ++i) { r[i] = lo[i]; r[8 + i] = hi[i]; }
    return r;
}
static __device__ __forceinline__ v8f wmma_bf16(v16bf a, v16bf b, v8f c) {
    return __builtin_amdgcn_wmma_f32_16x16x32_bf16(false, a, false, b, (short)0, c, false, false);
}

#if USE_TDM
// Issue a TDM 2-D bf16 tile load (Tensor Data Mover, VIMAGE TENSOR_LOAD_TO_LDS).
// Pads LDS rows: 64 elements (32 DWORDs) written, then +4 DWORDs skip -> stride 72.
static __device__ __forceinline__ void tdm_load_tile_bf16(
    unsigned lds_off, const bf16_t* gptr,
    unsigned dim0, unsigned dim1, unsigned stride0,
    unsigned tile0, unsigned tile1) {
    unsigned long long ga = (unsigned long long)(uintptr_t)gptr;
    u32x4 g0;
    g0[0] = 1u;                                        // count=1, user mode
    g0[1] = lds_off;                                   // lds_addr (bytes)
    g0[2] = (unsigned)(ga & 0xFFFFFFFFu);              // global_addr[31:0]
    g0[3] = (unsigned)((ga >> 32) & 0x01FFFFFFu) | (2u << 30);  // addr[56:32], type=2
    u32x8 g1;
    g1[0] = (1u << 16)        // data_size = 2 bytes
          | (1u << 20)        // pad_enable
          | (4u << 22)        // pad_interval: 32 DWORDs (one 64-elem row)
          | (3u << 25);       // pad_amount: 4 DWORDs (8 bf16)
    g1[1] = (dim0 & 0xFFFFu) << 16;                              // tensor_dim0[15:0]
    g1[2] = ((dim0 >> 16) & 0xFFFFu) | ((dim1 & 0xFFFFu) << 16); // dim0[31:16], dim1[15:0]
    g1[3] = ((dim1 >> 16) & 0xFFFFu) | ((tile0 & 0xFFFFu) << 16);// dim1[31:16], tile_dim0
    g1[4] = (tile1 & 0xFFFFu);                                   // tile_dim1, tile_dim2=0
    g1[5] = stride0;                                             // tensor_dim0_stride[31:0]
    g1[6] = 0u;                                                  // stride0[47:32], dim1_stride lo
    g1[7] = 0u;                                                  // dim1_stride hi (2-D tile)
    asm volatile("tensor_load_to_lds %0, %1" :: "s"(g0), "s"(g1) : "memory");
}
#endif

// ---------------- fp32 -> bf16 ----------------
__global__ __launch_bounds__(256) void cvt_bf16_kernel(const float* __restrict__ x,
                                                       bf16_t* __restrict__ y) {
    int i = (blockIdx.x * 256 + threadIdx.x) * 4;
    float4 v = *reinterpret_cast<const float4*>(x + i);
    ushort4 o;
    o.x = f2bf_bits(v.x); o.y = f2bf_bits(v.y); o.z = f2bf_bits(v.z); o.w = f2bf_bits(v.w);
    *reinterpret_cast<ushort4*>(reinterpret_cast<unsigned short*>(y) + i) = o;
}

// ---------------- W[K][N] fp32 -> Wt[N][K] bf16 ----------------
__global__ __launch_bounds__(256) void transpose_cvt_kernel(const float* __restrict__ W,
                                                            bf16_t* __restrict__ Wt) {
    __shared__ float tile[32][33];
    int k0 = blockIdx.x * 32, n0 = blockIdx.y * 32;
    for (int i = threadIdx.y; i < 32; i += 8)
        tile[i][threadIdx.x] = W[(size_t)(k0 + i) * DMODEL + n0 + threadIdx.x];
    __syncthreads();
    for (int i = threadIdx.y; i < 32; i += 8)
        Wt[(size_t)(n0 + i) * DMODEL + k0 + threadIdx.x] = f2bf(tile[threadIdx.x][i]);
}

// ---------------- GEMM: Y = A[M,1024] * Wt^T + bias ----------------
// Block: 128x128 tile, 8 waves, each wave 32x64 (2 m-frags x 4 n-frags).
// Register-staged double-buffered LDS pipeline.
// MODE 0: fp32 out[M][1024]; MODE 1: bf16 [B][H][T][dk]; MODE 2: bf16 [B][H][dk][T]
template <int MODE>
__global__ __launch_bounds__(256) void gemm_bf16_kernel(const bf16_t* __restrict__ A,
                                                        const bf16_t* __restrict__ Bt,
                                                        const float* __restrict__ bias,
                                                        void* __restrict__ Out) {
    __shared__ bf16_t As[2][128][40];
    __shared__ bf16_t Bs[2][128][40];
    const int tid  = threadIdx.x;
    const int wave = tid >> 5, lane = tid & 31;
    const int row  = lane & 15;
    const int kb   = (lane < 16) ? 0 : 8;    // A-frag K split
    const int klo  = (lane < 16) ? 0 : 16;   // unused; B-frags use kb too
    const int moff = (lane < 16) ? 0 : 8;
    const int wm = wave >> 1, wn = wave & 1;
    const int m0 = blockIdx.x * 128;
    const int n0 = blockIdx.y * 128;

    // each thread stages 2 A-chunks + 2 B-chunks (16B each) per k-step
    const int ldr0 = tid >> 1;                 // rows 0..127
    const int ldc0 = (tid & 1) * 16;           // col 0 or 16
    v8bf ra[2][2], rb[2][2];

    auto glb_load = [&](int k0) {
#pragma unroll
        for (int h = 0; h < 2; ++h) {
            ra[h][0] = *reinterpret_cast<const v8bf*>(&A [(size_t)(m0 + ldr0) * DMODEL + k0 + ldc0 + 8 * h]);
            rb[h][0] = *reinterpret_cast<const v8bf*>(&Bt[(size_t)(n0 + ldr0) * DMODEL + k0 + ldc0 + 8 * h]);
        }
        (void)ra; (void)rb;
    };
    auto lds_store = [&](int buf) {
#pragma unroll
        for (int h = 0; h < 2; ++h) {
            *reinterpret_cast<v8bf*>(&As[buf][ldr0][ldc0 + 8 * h]) = ra[h][0];
            *reinterpret_cast<v8bf*>(&Bs[buf][ldr0][ldc0 + 8 * h]) = rb[h][0];
        }
    };

    v8f acc[2][4] = {};
    glb_load(0);
    lds_store(0);
    for (int k0 = 0, it = 0; k0 < DMODEL; k0 += 32, ++it) {
        const int buf = it & 1;
        __syncthreads();
        const bool more = (k0 + 32 < DMODEL);
        if (more) glb_load(k0 + 32);                          // in flight during WMMAs
        if (k0 + 64 < DMODEL)
            __builtin_prefetch(&A[(size_t)(m0 + ldr0) * DMODEL + k0 + 64], 0, 0);
        v16bf a0 = load_frag(&As[buf][32 * wm + row][kb],      &As[buf][32 * wm + row][kb + 16]);
        v16bf a1 = load_frag(&As[buf][32 * wm + 16 + row][kb], &As[buf][32 * wm + 16 + row][kb + 16]);
#pragma unroll
        for (int j = 0; j < 4; ++j) {
            v16bf b = load_frag(&Bs[buf][64 * wn + 16 * j + row][kb],
                                &Bs[buf][64 * wn + 16 * j + row][kb + 16]);
            acc[0][j] = wmma_bf16(a0, b, acc[0][j]);
            acc[1][j] = wmma_bf16(a1, b, acc[1][j]);
        }
        if (more) lds_store(buf ^ 1);
    }
#pragma unroll
    for (int mt = 0; mt < 2; ++mt)
#pragma unroll
        for (int j = 0; j < 4; ++j) {
            int n = n0 + 64 * wn + 16 * j + row;
            float bv = bias[n];
#pragma unroll
            for (int r = 0; r < 8; ++r) {
                int m = m0 + 32 * wm + 16 * mt + r + moff;
                float val = acc[mt][j][r] + bv;
                if (MODE == 0) {
                    ((float*)Out)[(size_t)m * DMODEL + n] = val;
                } else {
                    int b_ = m >> 11, t = m & (SEQ - 1);
                    int h  = n >> 6,  d = n & (DK - 1);
                    if (MODE == 1)
                        ((bf16_t*)Out)[(((size_t)(b_ * NHEADS + h)) * SEQ + t) * DK + d] = f2bf(val);
                    else
                        ((bf16_t*)Out)[(((size_t)(b_ * NHEADS + h)) * DK + d) * SEQ + t] = f2bf(val);
                }
            }
        }
}

// ---------------- Flash attention ----------------
// Q[BH][T][dk], K[BH][T][dk], Vt[BH][dk][T] -> ctx bf16 [B][T][D]
// K/V tiles DMA'd by the Tensor Data Mover (double buffered), P staged per-wave in LDS.
__global__ __launch_bounds__(256) void attention_kernel(const bf16_t* __restrict__ Q,
                                                        const bf16_t* __restrict__ K,
                                                        const bf16_t* __restrict__ Vt,
                                                        bf16_t* __restrict__ ctx) {
    // Manual LDS partition (single block => known offsets for TDM descriptors).
    // Ks: 2 x [64][72] bf16 (9216 B each), Vs: same, Ps: 8 x [16][72] bf16.
    __shared__ __align__(16) unsigned char smem[2 * 9216 + 2 * 9216 + 18432];
    bf16_t (*Ks)[64][72] = reinterpret_cast<bf16_t(*)[64][72]>(smem);
    bf16_t (*Vs)[64][72] = reinterpret_cast<bf16_t(*)[64][72]>(smem + 2 * 9216);
    bf16_t (*Ps)[16][72] = reinterpret_cast<bf16_t(*)[16][72]>(smem + 4 * 9216);

    const int tid  = threadIdx.x;
    const int wave = tid >> 5, lane = tid & 31;
    const int row  = lane & 15;
    const int kb   = (lane < 16) ? 0 : 8;
    const int klo  = (lane < 16) ? 0 : 16;
    const int moff = (lane < 16) ? 0 : 8;
    const int bh    = blockIdx.y;
    const int qtile = blockIdx.x;

    const bf16_t* Qh = Q  + (size_t)bh * SEQ * DK;
    const bf16_t* Kh = K  + (size_t)bh * SEQ * DK;
    const bf16_t* Vh = Vt + (size_t)bh * DK * SEQ;

#if USE_TDM
    const unsigned ks_off = (unsigned)(uintptr_t)&Ks[0][0][0];  // LDS flat addr low bits = offset
    const unsigned vs_off = (unsigned)(uintptr_t)&Vs[0][0][0];
#endif

    const int qrow = qtile * 128 + wave * 16 + row;
    v16bf aq0 = load_frag(&Qh[(size_t)qrow * DK + kb],      &Qh[(size_t)qrow * DK + kb + 16]);
    v16bf aq1 = load_frag(&Qh[(size_t)qrow * DK + 32 + kb], &Qh[(size_t)qrow * DK + 32 + kb + 16]);

    v8f o[4] = {};
    float mrow[8], lrow[8];
#pragma unroll
    for (int r = 0; r < 8; ++r) { mrow[r] = -3.0e38f; lrow[r] = 0.0f; }
    const float scale = 0.125f;  // 1/sqrt(64)

#if USE_TDM
    if (wave == 0) {  // TDM ignores EXEC; issue once per workgroup (wave 0)
        tdm_load_tile_bf16(ks_off, &Kh[0], DK, SEQ, DK, 64, 64);
        tdm_load_tile_bf16(vs_off, &Vh[0], SEQ, DK, SEQ, 64, 64);
    }
#endif

    const int nstage = SEQ / 64;
    for (int it = 0; it < nstage; ++it) {
        const int buf = it & 1;
        const int kv0 = it * 64;
#if USE_TDM
        if (wave == 0) __builtin_amdgcn_s_wait_tensorcnt(0);
        __syncthreads();
        if (wave == 0 && it + 1 < nstage) {
            const int kv1 = kv0 + 64;
            tdm_load_tile_bf16(ks_off + (buf ^ 1) * 9216, &Kh[(size_t)kv1 * DK], DK, SEQ, DK, 64, 64);
            tdm_load_tile_bf16(vs_off + (buf ^ 1) * 9216, &Vh[kv1], SEQ, DK, SEQ, 64, 64);
        }
#else
        __syncthreads();
#pragma unroll
        for (int p = 0; p < 2; ++p) {
            int idx = p * 256 + tid;
            int r = idx >> 3, c = (idx & 7) * 8;
            *reinterpret_cast<v8bf*>(&Ks[buf][r][c]) =
                *reinterpret_cast<const v8bf*>(&Kh[(size_t)(kv0 + r) * DK + c]);
            *reinterpret_cast<v8bf*>(&Vs[buf][r][c]) =
                *reinterpret_cast<const v8bf*>(&Vh[(size_t)r * SEQ + kv0 + c]);
        }
        __syncthreads();
#endif

        // S = Q * K^T  (4 kv n-tiles x 2 k-steps over dk)
        v8f s[4] = {};
#pragma unroll
        for (int j = 0; j < 4; ++j) {
            v16bf bk0 = load_frag(&Ks[buf][16 * j + row][kb],      &Ks[buf][16 * j + row][kb + 16]);
            v16bf bk1 = load_frag(&Ks[buf][16 * j + row][32 + kb], &Ks[buf][16 * j + row][32 + kb + 16]);
            s[j] = wmma_bf16(aq0, bk0, s[j]);
            s[j] = wmma_bf16(aq1, bk1, s[j]);
        }
#pragma unroll
        for (int j = 0; j < 4; ++j)
#pragma unroll
            for (int r = 0; r < 8; ++r) s[j][r] *= scale;

        // online softmax (row = 16 lanes of this half)
        float newm[8], rs[8];
#pragma unroll
        for (int r = 0; r < 8; ++r) {
            float mx = s[0][r];
#pragma unroll
            for (int j = 1; j < 4; ++j) mx = fmaxf(mx, s[j][r]);
            mx = fmaxf(mx, __shfl_xor(mx, 1));
            mx = fmaxf(mx, __shfl_xor(mx, 2));
            mx = fmaxf(mx, __shfl_xor(mx, 4));
            mx = fmaxf(mx, __shfl_xor(mx, 8));
            newm[r] = fmaxf(mrow[r], mx);
            rs[r] = 0.0f;
        }
#pragma unroll
        for (int j = 0; j < 4; ++j)
#pragma unroll
            for (int r = 0; r < 8; ++r) {
                float p = __expf(s[j][r] - newm[r]);
                s[j][r] = p;
                rs[r] += p;
            }
#pragma unroll
        for (int r = 0; r < 8; ++r) {
            rs[r] += __shfl_xor(rs[r], 1);
            rs[r] += __shfl_xor(rs[r], 2);
            rs[r] += __shfl_xor(rs[r], 4);
            rs[r] += __shfl_xor(rs[r], 8);
            float cf = __expf(mrow[r] - newm[r]);
            lrow[r] = lrow[r] * cf + rs[r];
            mrow[r] = newm[r];
#pragma unroll
            for (int j = 0; j < 4; ++j) o[j][r] *= cf;
        }
        // stage P: C-layout -> A-layout via per-wave LDS tile
#pragma unroll
        for (int j = 0; j < 4; ++j)
#pragma unroll
            for (int r = 0; r < 8; ++r)
                Ps[wave][r + moff][16 * j + row] = f2bf(s[j][r]);

        // ctx += P * V  (2 kv k-steps x 4 dk n-tiles)
#pragma unroll
        for (int ks = 0; ks < 2; ++ks) {
            v16bf pa = load_frag(&Ps[wave][row][32 * ks + kb], &Ps[wave][row][32 * ks + kb + 16]);
#pragma unroll
            for (int j = 0; j < 4; ++j) {
                v16bf bv = load_frag(&Vs[buf][16 * j + row][32 * ks + kb],
                                     &Vs[buf][16 * j + row][32 * ks + kb + 16]);
                o[j] = wmma_bf16(pa, bv, o[j]);
            }
        }
    }
    // normalize + store ctx bf16 [B][T][D]
    const int b_ = bh >> 4, h = bh & (NHEADS - 1);
#pragma unroll
    for (int j = 0; j < 4; ++j)
#pragma unroll
        for (int r = 0; r < 8; ++r) {
            int m = qtile * 128 + wave * 16 + r + moff;
            int d = 16 * j + row;
            ctx[((size_t)b_ * SEQ + m) * DMODEL + h * DK + d] = f2bf(o[j][r] / lrow[r]);
        }
}

extern "C" void kernel_launch(void* const* d_in, const int* in_sizes, int n_in,
                              void* d_out, int out_size, void* d_ws, size_t ws_size,
                              hipStream_t stream) {
    const float* x  = (const float*)d_in[0];
    const float* Wq = (const float*)d_in[1];
    const float* bq = (const float*)d_in[2];
    const float* Wk = (const float*)d_in[3];
    const float* bk = (const float*)d_in[4];
    const float* Wv = (const float*)d_in[5];
    const float* bv = (const float*)d_in[6];
    const float* Wo = (const float*)d_in[7];
    const float* bo = (const float*)d_in[8];
    float* out = (float*)d_out;

    char* ws = (char*)d_ws;
    const size_t XB   = (size_t)MROWS * DMODEL * 2;
    const size_t WT   = (size_t)DMODEL * DMODEL * 2;
    const size_t HEAD = (size_t)BATCH * NHEADS * SEQ * DK * 2;
    bf16_t* xb   = (bf16_t*)(ws);
    bf16_t* Wqt  = (bf16_t*)(ws + XB);
    bf16_t* Wkt  = (bf16_t*)(ws + XB + WT);
    bf16_t* Wvt  = (bf16_t*)(ws + XB + 2 * WT);
    bf16_t* Wot  = (bf16_t*)(ws + XB + 3 * WT);
    bf16_t* Qh   = (bf16_t*)(ws + XB + 4 * WT);
    bf16_t* Kh   = (bf16_t*)(ws + XB + 4 * WT + HEAD);
    bf16_t* Vth  = (bf16_t*)(ws + XB + 4 * WT + 2 * HEAD);
    bf16_t* ctxb = (bf16_t*)(ws + XB + 4 * WT + 3 * HEAD);

    cvt_bf16_kernel<<<(MROWS * DMODEL) / (256 * 4), 256, 0, stream>>>(x, xb);
    dim3 tb(32, 8), tg(32, 32);
    transpose_cvt_kernel<<<tg, tb, 0, stream>>>(Wq, Wqt);
    transpose_cvt_kernel<<<tg, tb, 0, stream>>>(Wk, Wkt);
    transpose_cvt_kernel<<<tg, tb, 0, stream>>>(Wv, Wvt);
    transpose_cvt_kernel<<<tg, tb, 0, stream>>>(Wo, Wot);

    dim3 gg(MROWS / 128, DMODEL / 128);
    gemm_bf16_kernel<1><<<gg, 256, 0, stream>>>(xb, Wqt, bq, Qh);
    gemm_bf16_kernel<1><<<gg, 256, 0, stream>>>(xb, Wkt, bk, Kh);
    gemm_bf16_kernel<2><<<gg, 256, 0, stream>>>(xb, Wvt, bv, Vth);

    dim3 ag(SEQ / 128, BATCH * NHEADS);
    attention_kernel<<<ag, 256, 0, stream>>>(Qh, Kh, Vth, ctxb);

    gemm_bf16_kernel<0><<<gg, 256, 0, stream>>>(ctxb, Wot, bo, out);
}
